// FusedExpertsWrapper_33122787787378
// MI455X (gfx1250) — matrix operand order repaired and checked by reading, not compiled, not run on hardware
//
#include <hip/hip_runtime.h>
#include <hip/hip_bf16.h>
#include <cstddef>

// ---------------------------------------------------------------------------
// GPT-OSS MoE FFN, MI455X (gfx1250). X=A*B=16 row groups, E=16 experts,
// M=128 rows, H=2048, I=2048, 2I=4096 interleaved gate/up.
// Output fp32 flat index ((x*128+m)*E + e)*H + h.
// Workspace: activated bf16 (x,e,m,i) = 16*16*128*2048*2B = 128 MB.
// ---------------------------------------------------------------------------

typedef __attribute__((ext_vector_type(16))) __bf16          bf16x16;
typedef __attribute__((ext_vector_type(8)))  float           f32x8;
typedef __attribute__((ext_vector_type(4)))  float           f32x4;
typedef __attribute__((ext_vector_type(8)))  unsigned short  u16x8;
typedef __attribute__((ext_vector_type(16))) unsigned short  u16x16;
typedef __attribute__((ext_vector_type(2)))  unsigned int    u32x2;
typedef __attribute__((ext_vector_type(4)))  unsigned int    u32x4;
typedef __attribute__((ext_vector_type(4)))  int             i32x4;

#if defined(__HIP_DEVICE_COMPILE__) && \
    __has_builtin(__builtin_amdgcn_global_load_async_to_lds_b128)
#define USE_ASYNC_LDS 1
typedef __attribute__((address_space(1))) i32x4 g_i32x4;   // global int4
typedef __attribute__((address_space(3))) i32x4 l_i32x4;   // LDS int4
#else
#define USE_ASYNC_LDS 0
#endif

__device__ __forceinline__ void wait_async_zero() {
#if USE_ASYNC_LDS
#if __has_builtin(__builtin_amdgcn_s_wait_asynccnt)
  __builtin_amdgcn_s_wait_asynccnt(0);
#else
  asm volatile("s_wait_asynccnt 0x0" ::: "memory");
#endif
#endif
}

__device__ __forceinline__ unsigned short f2bf(float f) {
  // round-to-nearest-even fp32 -> bf16
  unsigned int u = __float_as_uint(f);
  u += 0x7FFFu + ((u >> 16) & 1u);
  return (unsigned short)(u >> 16);
}
__device__ __forceinline__ unsigned int pack2bf(float a, float b) {
  return (unsigned int)f2bf(a) | ((unsigned int)f2bf(b) << 16);
}

// A fragment (16x32 bf16, MxK). ISA 7.12.2:
//  lane<16 : row M=lane,    K = {0..7, 16..23}
//  lane>=16: row M=lane-16, K = {8..15, 24..31}
__device__ __forceinline__ bf16x16
load_a_frag(const unsigned short (*As)[40], int rowbase, int lane) {
  const int row = rowbase + (lane & 15);
  const int kh  = (lane >> 4) * 8;
  const u16x8 lo = *(const u16x8*)&As[row][kh];
  const u16x8 hi = *(const u16x8*)&As[row][16 + kh];
  u16x16 v;
#pragma unroll
  for (int i = 0; i < 8; ++i) { v[i] = lo[i]; v[i + 8] = hi[i]; }
  return __builtin_bit_cast(bf16x16, v);
}

// B fragment (32x16 bf16, KxN), LDS N-major ([col][k], stride 40):
//  lane<16: col N=lane, K={0..7,16..23};  lane>=16: col N=lane-16, K={8..15,24..31}
__device__ __forceinline__ bf16x16
load_b_frag(const unsigned short (*Bs)[40], int colbase, int lane) {
  const int col = colbase + (lane & 15);
  const int kh  = (lane >> 4) * 8;
  const u16x8 lo = *(const u16x8*)&Bs[col][kh];
  const u16x8 hi = *(const u16x8*)&Bs[col][16 + kh];
  u16x16 v;
#pragma unroll
  for (int i = 0; i < 8; ++i) { v[i] = lo[i]; v[i + 8] = hi[i]; }
  return __builtin_bit_cast(bf16x16, v);
}

__device__ __forceinline__ f32x8
wmma_bf16(bf16x16 a, bf16x16 b, f32x8 c) {
  return __builtin_amdgcn_wmma_f32_16x16x32_bf16(false, a, false, b,
                                                 (short)0, c, false, false);
}

// ---------------------------------------------------------------------------
// Kernel 1: gate_up GEMM + bias + activation -> activated bf16 (workspace)
// block = (e, x, 64-wide activated col tile); 8 waves: 4 row x 2 col;
// each wave owns 32 rows x (32 gate + 32 up) so activation is lane-local.
// ---------------------------------------------------------------------------
__global__ __launch_bounds__(256) void moe_gemm1_act(
    const float* __restrict__ disp,       // (X*E*M, H)
    const float* __restrict__ gup,        // (E, H, 2I)
    const float* __restrict__ gbias,      // (E, 2I)
    unsigned short* __restrict__ act)     // (X, E, M, I) bf16
{
  __shared__ __align__(16) unsigned short Sh[2][128][40];
  unsigned short (*As)[40] = Sh[0];
  unsigned short (*Bs)[40] = Sh[1];

  const int bid = blockIdx.x;
  const int nt  = bid & 31;
  const int x   = (bid >> 5) & 15;
  const int e   = bid >> 9;
  const int ci0 = nt * 64;

  const int tid  = threadIdx.x;
  const int lane = tid & 31;
  const int wid  = tid >> 5;
  const int wr   = wid & 3;
  const int wc   = wid >> 2;

  const float* Ab = disp + (size_t)(x * 16 + e) * 128 * 2048;
  const float* Bb = gup  + (size_t)e * 2048 * 4096 + 2 * ci0;
  const float* bb = gbias + (size_t)e * 4096;

  f32x8 accg[2][2] = {};
  f32x8 accu[2][2] = {};

  for (int k0 = 0; k0 < 2048; k0 += 32) {
    // ---- phase 1: batched vector loads (no per-element branches) ----------
    f32x4 va[4], vb[4];
#pragma unroll
    for (int i = 0; i < 4; ++i) {        // A: 128 rows x 32 K = 1024 float4
      const int idx = tid + 256 * i;
      va[i] = *(const f32x4*)(Ab + (size_t)(idx >> 3) * 2048 + k0 + (idx & 7) * 4);
    }
#pragma unroll
    for (int i = 0; i < 4; ++i) {        // B: 32 K x 128 cols = 1024 float4
      const int idx = tid + 256 * i;
      vb[i] = *(const f32x4*)(Bb + (size_t)(k0 + (idx >> 5)) * 4096 + (idx & 31) * 4);
    }
    if (k0 + 32 < 2048) {                // uniform branch: prefetch weights
#pragma unroll
      for (int i = 0; i < 4; ++i) {
        const int idx = tid + 256 * i;
        __builtin_prefetch(Bb + (size_t)(k0 + 32 + (idx >> 5)) * 4096 + (idx & 31) * 4, 0, 3);
      }
    }
    // ---- phase 2: convert + LDS store -------------------------------------
#pragma unroll
    for (int i = 0; i < 4; ++i) {        // A: packed 2xbf16 dwords, 8B stores
      const int idx = tid + 256 * i;
      const int row = idx >> 3, c = (idx & 7) * 4;
      u32x2 p;
      p.x = pack2bf(va[i].x, va[i].y);
      p.y = pack2bf(va[i].z, va[i].w);
      *(u32x2*)&As[row][c] = p;
    }
#pragma unroll
    for (int i = 0; i < 4; ++i) {        // B: de-interleave + transpose
      const int idx = tid + 256 * i;
      const int row = idx >> 5, c = (idx & 31) * 4, g = c >> 1;
      Bs[g][row]          = f2bf(vb[i].x);   // gate -> cols 0..63
      Bs[64 + g][row]     = f2bf(vb[i].y);   // up   -> cols 64..127
      Bs[g + 1][row]      = f2bf(vb[i].z);
      Bs[64 + g + 1][row] = f2bf(vb[i].w);
    }
    __syncthreads();

    // ---- 8 WMMAs ----------------------------------------------------------
    bf16x16 af[2];
#pragma unroll
    for (int t = 0; t < 2; ++t) af[t] = load_a_frag(As, 32 * wr + 16 * t, lane);
#pragma unroll
    for (int u = 0; u < 2; ++u) {
      const bf16x16 bg = load_b_frag(Bs,      32 * wc + 16 * u, lane);
      const bf16x16 bu = load_b_frag(Bs, 64 + 32 * wc + 16 * u, lane);
#pragma unroll
      for (int t = 0; t < 2; ++t) {
        accg[t][u] = wmma_bf16(af[t], bg, accg[t][u]);
        accu[t][u] = wmma_bf16(af[t], bu, accu[t][u]);
      }
    }
    __syncthreads();
  }

  // ---- epilogue: bias + clamp + SiLU-glu, bounce through LDS, coalesced ----
  unsigned short* Es = &Sh[0][0][0];     // 128 x 64 bf16 tile (8192 <= 10240)
  const int nloc  = lane & 15;
  const int rhalf = (lane >> 4) * 8;
#pragma unroll
  for (int t = 0; t < 2; ++t) {
#pragma unroll
    for (int u = 0; u < 2; ++u) {
      const int jl = 32 * wc + 16 * u + nloc;   // tile-local activated col
      const int j  = ci0 + jl;
      const float gb = bb[2 * j];
      const float ub = bb[2 * j + 1];
#pragma unroll
      for (int r = 0; r < 8; ++r) {
        const int m = 32 * wr + 16 * t + rhalf + r;
        float g  = accg[t][u][r] + gb;
        float up = accu[t][u][r] + ub;
        g  = fminf(g, 7.0f);
        up = fminf(fmaxf(up, -7.0f), 7.0f);
        const float glu = g / (1.0f + __expf(-1.702f * g));
        Es[m * 64 + jl] = f2bf((up + 1.0f) * glu);
      }
    }
  }
  __syncthreads();
  unsigned short* ob = act + (size_t)(x * 16 + e) * 128 * 2048;
#pragma unroll
  for (int i = 0; i < 4; ++i) {          // 1024 x 16B chunks, coalesced
    const int idx = tid + 256 * i;
    const int row = idx >> 3, c8 = (idx & 7) * 8;
    *(u32x4*)(ob + (size_t)row * 2048 + ci0 + c8) =
        *(const u32x4*)(Es + row * 64 + c8);
  }
}

// ---------------------------------------------------------------------------
// Kernel 2: activated(bf16) x down_proj + bias -> out fp32 (transposed store)
// A tile is already bf16: stage it with GLOBAL_LOAD_ASYNC_TO_LDS_B128.
// ---------------------------------------------------------------------------
__global__ __launch_bounds__(256) void moe_gemm2_out(
    const unsigned short* __restrict__ act,  // (X, E, M, I) bf16
    const float* __restrict__ down,          // (E, I, H)
    const float* __restrict__ dbias,         // (E, H)
    float* __restrict__ out)
{
  __shared__ __align__(16) unsigned short Sh[2][128][40];
  unsigned short (*As)[40] = Sh[0];
  unsigned short (*Bs)[40] = Sh[1];

  const int bid = blockIdx.x;
  const int ht  = bid & 15;
  const int x   = (bid >> 4) & 15;
  const int e   = bid >> 8;
  const int h0  = ht * 128;

  const int tid  = threadIdx.x;
  const int lane = tid & 31;
  const int wid  = tid >> 5;
  const int wr   = wid & 3;
  const int wc   = wid >> 2;

  const unsigned short* Ab = act + (size_t)(x * 16 + e) * 128 * 2048;
  const float* Bb = down + (size_t)e * 2048 * 2048 + h0;

  f32x8 acc[2][4] = {};

  for (int k0 = 0; k0 < 2048; k0 += 32) {
    // ---- A tile: pure bf16 copy global -> LDS via async DMA ---------------
#pragma unroll
    for (int i = 0; i < 2; ++i) {        // 512 x 16B chunks
      const int idx = tid + 256 * i;
      const int row = idx >> 2, c8 = (idx & 3) * 8;
      const unsigned short* src = Ab + (size_t)row * 2048 + k0 + c8;
      unsigned short* dst = &As[row][c8];
#if USE_ASYNC_LDS
      __builtin_amdgcn_global_load_async_to_lds_b128(
          (g_i32x4*)src, (l_i32x4*)dst, 0, 0);
#else
      *(u32x4*)dst = *(const u32x4*)src;
#endif
    }
    // ---- B tile: fp32 -> bf16, transposed to N-major ----------------------
    f32x4 vb[4];
#pragma unroll
    for (int i = 0; i < 4; ++i) {
      const int idx = tid + 256 * i;
      vb[i] = *(const f32x4*)(Bb + (size_t)(k0 + (idx >> 5)) * 2048 + (idx & 31) * 4);
    }
    if (k0 + 32 < 2048) {
#pragma unroll
      for (int i = 0; i < 4; ++i) {
        const int idx = tid + 256 * i;
        __builtin_prefetch(Bb + (size_t)(k0 + 32 + (idx >> 5)) * 2048 + (idx & 31) * 4, 0, 3);
      }
    }
#pragma unroll
    for (int i = 0; i < 4; ++i) {
      const int idx = tid + 256 * i;
      const int row = idx >> 5, c = (idx & 31) * 4;
      Bs[c][row]     = f2bf(vb[i].x);
      Bs[c + 1][row] = f2bf(vb[i].y);
      Bs[c + 2][row] = f2bf(vb[i].z);
      Bs[c + 3][row] = f2bf(vb[i].w);
    }
    wait_async_zero();
    __syncthreads();

    // ---- 8 WMMAs ----------------------------------------------------------
    bf16x16 af[2];
#pragma unroll
    for (int t = 0; t < 2; ++t) af[t] = load_a_frag(As, 32 * wr + 16 * t, lane);
#pragma unroll
    for (int u = 0; u < 4; ++u) {
      const bf16x16 bf = load_b_frag(Bs, 64 * wc + 16 * u, lane);
#pragma unroll
      for (int t = 0; t < 2; ++t) acc[t][u] = wmma_bf16(af[t], bf, acc[t][u]);
    }
    __syncthreads();
  }

  // ---- epilogue: + bias, transposed store (x,m,e,h) -------------------------
  const int nloc  = lane & 15;
  const int rhalf = (lane >> 4) * 8;
  const float* bb = dbias + (size_t)e * 2048;
#pragma unroll
  for (int t = 0; t < 2; ++t) {
#pragma unroll
    for (int u = 0; u < 4; ++u) {
      const int h = h0 + 64 * wc + 16 * u + nloc;
      const float hb = bb[h];
#pragma unroll
      for (int r = 0; r < 8; ++r) {
        const int m = 32 * wr + 16 * t + rhalf + r;
        out[((size_t)(x * 128 + m) * 16 + e) * 2048 + h] = acc[t][u][r] + hb;
      }
    }
  }
}

// ---------------------------------------------------------------------------
extern "C" void kernel_launch(void* const* d_in, const int* in_sizes, int n_in,
                              void* d_out, int out_size, void* d_ws, size_t ws_size,
                              hipStream_t stream) {
  (void)in_sizes; (void)n_in; (void)out_size; (void)ws_size;
  const float* disp  = (const float*)d_in[0];
  const float* gup   = (const float*)d_in[1];
  const float* gbias = (const float*)d_in[2];
  const float* down  = (const float*)d_in[3];
  const float* dbias = (const float*)d_in[4];
  // d_in[5] sparsity_remap, d_in[6] BD, d_in[7] S: unused by the reference math
  float* out = (float*)d_out;
  unsigned short* act = (unsigned short*)d_ws;   // needs 128 MB

  moe_gemm1_act<<<16 * 16 * 32, 256, 0, stream>>>(disp, gup, gbias, act);
  moe_gemm2_out<<<16 * 16 * 16, 256, 0, stream>>>(act, down, dbias, out);
}
